// NnclrAllswapLoss_43843026157756
// MI455X (gfx1250) — matrix operand order repaired and checked by reading, not compiled, not run on hardware
//
#include <hip/hip_runtime.h>
#include <hip/hip_bf16.h>

typedef __bf16 bf16;
typedef __attribute__((ext_vector_type(16))) __bf16 v16bf;
typedef __attribute__((ext_vector_type(8)))  __bf16 bf16x8;
typedef __attribute__((ext_vector_type(8)))  float  v8f;

#define NB    2048      // batch
#define ND    256       // feature dim
#define NT    (NB / 16) // column tiles per row sweep
#define INV_T 5.0f      // 1/temperature, folded into P
#define SHIFT 5.0f      // max possible logit (unit vectors / 0.2)

// ---------------------------------------------------------------------------
// Kernel 1: L2-normalize rows, convert to bf16 into workspace.
//   pn: [i(2)][b(2048)][256] bf16  (scaled by 1/T)
//   qn: [j(2)][b(2048)][256] bf16  (only predicted views 0,1 matter)
// Also zero the 4 per-pair loss accumulators. One wave per row.
// ---------------------------------------------------------------------------
__global__ __launch_bounds__(256) void nnclr_normalize_kernel(
    const float* __restrict__ projected, const float* __restrict__ predicted,
    bf16* __restrict__ pn, bf16* __restrict__ qn, float* __restrict__ acc)
{
  if (blockIdx.x == 0 && threadIdx.x < 4) acc[threadIdx.x] = 0.0f;

  const int lane = threadIdx.x & 31;
  const int w    = blockIdx.x * 8 + (threadIdx.x >> 5);   // 0..8191

  const float* src;
  bf16* dst;
  float extra;
  if (w < 2 * NB) {                    // projected view i, row b
    int i = w >> 11, b = w & (NB - 1);
    src   = projected + ((size_t)b * 2 + i) * ND;
    dst   = pn + (size_t)w * ND;       // w == i*NB + b
    extra = INV_T;
  } else {                             // predicted view j in {0,1}, row b
    int wq = w - 2 * NB;
    int j = wq >> 11, b = wq & (NB - 1);
    src   = predicted + ((size_t)b * 4 + j) * ND;
    dst   = qn + (size_t)wq * ND;
    extra = 1.0f;
  }

  float4 x0 = ((const float4*)src)[lane * 2 + 0];
  float4 x1 = ((const float4*)src)[lane * 2 + 1];
  float ss = x0.x*x0.x + x0.y*x0.y + x0.z*x0.z + x0.w*x0.w
           + x1.x*x1.x + x1.y*x1.y + x1.z*x1.z + x1.w*x1.w;
  #pragma unroll
  for (int o = 16; o > 0; o >>= 1) ss += __shfl_xor(ss, o, 32);

  const float sc = extra / fmaxf(sqrtf(ss), 1e-12f);

  bf16x8 y;
  y[0]=(bf16)(x0.x*sc); y[1]=(bf16)(x0.y*sc); y[2]=(bf16)(x0.z*sc); y[3]=(bf16)(x0.w*sc);
  y[4]=(bf16)(x1.x*sc); y[5]=(bf16)(x1.y*sc); y[6]=(bf16)(x1.z*sc); y[7]=(bf16)(x1.w*sc);
  ((bf16x8*)dst)[lane] = y;
}

// ---------------------------------------------------------------------------
// Helpers for the fused GEMM+LSE kernel.
// ---------------------------------------------------------------------------
__device__ __forceinline__ void load_btile(const bf16* __restrict__ brow,
                                           int koffB, v16bf (&bb)[8])
{
  // B layout: lane = column; lane<16 holds K 0..7,16..23; lane>=16 holds K 8..15,24..31
  #pragma unroll
  for (int kc = 0; kc < 8; ++kc) {
    bf16x8 lo = ((const bf16x8*)brow)[(kc * 32 + koffB)     >> 3];
    bf16x8 hi = ((const bf16x8*)brow)[(kc * 32 + koffB + 16) >> 3];
    #pragma unroll
    for (int e = 0; e < 8; ++e) { bb[kc][e] = lo[e]; bb[kc][e + 8] = hi[e]; }
  }
}

__device__ __forceinline__ void tile_update(const v16bf (&a)[8],
                                            const v16bf (&bb)[8],
                                            int t, int rowTile, int lane,
                                            float (&s)[8], float& diagv)
{
  v8f c = {0.f, 0.f, 0.f, 0.f, 0.f, 0.f, 0.f, 0.f};
  #pragma unroll
  for (int kc = 0; kc < 8; ++kc)
    c = __builtin_amdgcn_wmma_f32_16x16x32_bf16(
            /*neg_a=*/false, a[kc], /*neg_b=*/false, bb[kc],
            /*c_mod=*/(short)0, c, /*reuse_a=*/false, /*reuse_b=*/false);

  // C layout: VGPR e -> row e + 8*(lane>=16), col = lane%16 (+ 16*t global)
  const bool dt = (t == rowTile);
  #pragma unroll
  for (int e = 0; e < 8; ++e) {
    float v = c[e];
    if (dt && ((lane < 8 && e == lane) || (lane >= 24 && e == lane - 24)))
      diagv = v;                          // S[b,b]
    s[e] += __expf(v - SHIFT);
  }
}

// ---------------------------------------------------------------------------
// Kernel 2: fused GEMM (bf16 WMMA, fp32 acc) + per-row logsumexp - diag.
// Grid: (128 row tiles, 4 view pairs), one wave32 per block (EXEC all ones).
// Double-buffered B tiles in registers so global_load_b128 for tile t+1
// overlaps the 8 WMMAs of tile t; global_prefetch warms tiles further out.
// ---------------------------------------------------------------------------
__global__ __launch_bounds__(32) void nnclr_lse_gemm_kernel(
    const bf16* __restrict__ pn, const bf16* __restrict__ qn,
    float* __restrict__ acc)
{
  const int lane    = threadIdx.x;
  const int pair    = blockIdx.y;            // i*2 + j
  const int i       = pair >> 1;
  const int j       = pair & 1;
  const int rowTile = blockIdx.x;            // 0..127
  const int rowBase = rowTile * 16;
  const int lrow    = lane & 15;

  // --- A fragments (persistent): 16x32 bf16 per K-chunk, ISA A-layout ---
  // lane<16:  elems 0-7 = K 0..7,  elems 8-15 = K 16..23  (of the chunk)
  // lane>=16: elems 0-7 = K 8..15, elems 8-15 = K 24..31
  const bf16* arow  = pn + ((size_t)i * NB + rowBase + lrow) * ND;
  const int   koffA = (lane < 16) ? 0 : 8;
  v16bf a[8];
  #pragma unroll
  for (int kc = 0; kc < 8; ++kc) {
    bf16x8 lo = ((const bf16x8*)arow)[(kc * 32 + koffA)      >> 3];
    bf16x8 hi = ((const bf16x8*)arow)[(kc * 32 + koffA + 16) >> 3];
    #pragma unroll
    for (int e = 0; e < 8; ++e) { a[kc][e] = lo[e]; a[kc][e + 8] = hi[e]; }
  }

  const bf16* qbase = qn + (size_t)j * NB * ND;
  const int   koffB = (lane < 16) ? 0 : 8;

  float s[8];
  #pragma unroll
  for (int e = 0; e < 8; ++e) s[e] = 0.0f;
  float diagv = 0.0f;

  // Software-pipelined sweep over 128 column tiles (two per iteration).
  v16bf bb0[8], bb1[8];
  load_btile(qbase + (size_t)lrow * ND, koffB, bb0);

  for (int t = 0; t < NT; t += 2) {
    load_btile(qbase + ((size_t)(t + 1) * 16 + lrow) * ND, koffB, bb1);
    {
      int tp = (t + 4 < NT) ? (t + 4) : (NT - 1);   // warm cache ahead
      __builtin_prefetch(qbase + ((size_t)tp * 16 + lrow) * ND, 0, 1);
    }
    tile_update(a, bb0, t, rowTile, lane, s, diagv);

    if (t + 2 < NT)
      load_btile(qbase + ((size_t)(t + 2) * 16 + lrow) * ND, koffB, bb0);
    tile_update(a, bb1, t + 1, rowTile, lane, s, diagv);
  }

  // Reduce exp-sums across the 16 lanes holding each row's columns.
  #pragma unroll
  for (int o = 1; o < 16; o <<= 1) {
    #pragma unroll
    for (int e = 0; e < 8; ++e) s[e] += __shfl_xor(s[e], o, 32);
  }

  float val = 0.0f;
  if (lane == 0 || lane == 16) {           // one lane per half owns 8 rows' LSE
    #pragma unroll
    for (int e = 0; e < 8; ++e) val += SHIFT + __logf(s[e]);
  }
  if (lane < 8 || lane >= 24) val -= diagv; // diag-owning lanes subtract

  #pragma unroll
  for (int o = 16; o > 0; o >>= 1) val += __shfl_xor(val, o, 32);
  if (lane == 0) atomicAdd(&acc[pair], val);
}

// ---------------------------------------------------------------------------
// Kernel 3: combine per-pair sums into (total, global_loss, local_loss).
// ---------------------------------------------------------------------------
__global__ void nnclr_finalize_kernel(const float* __restrict__ acc,
                                      float* __restrict__ out)
{
  if (blockIdx.x == 0 && threadIdx.x == 0) {
    const float inv = 1.0f / (float)NB;
    float L00 = acc[0] * inv, L01 = acc[1] * inv;
    float L10 = acc[2] * inv, L11 = acc[3] * inv;
    float g = L01 + L10;                       // off-diagonal, j < Vg
    float l = L00 + L01 + L10 + L11;           // j < Vl (= 2)
    out[0] = (g + l) / 6.0f;                   // (2 global + 4 local terms)
    out[1] = g * 0.5f;
    out[2] = l * 0.25f;
  }
}

extern "C" void kernel_launch(void* const* d_in, const int* in_sizes, int n_in,
                              void* d_out, int out_size, void* d_ws, size_t ws_size,
                              hipStream_t stream) {
  const float* projected = (const float*)d_in[0];   // [2048, 2, 256] f32
  const float* predicted = (const float*)d_in[1];   // [2048, 4, 256] f32

  bf16*  pn  = (bf16*)d_ws;                                     // 2 MiB
  bf16*  qn  = pn + (size_t)2 * NB * ND;                        // 2 MiB
  float* acc = (float*)((char*)d_ws + (size_t)4 * NB * ND * sizeof(bf16)); // 16 B
  float* out = (float*)d_out;                                   // 3 floats

  nnclr_normalize_kernel<<<dim3(1024), dim3(256), 0, stream>>>(
      projected, predicted, pn, qn, acc);
  nnclr_lse_gemm_kernel<<<dim3(128, 4), dim3(32), 0, stream>>>(pn, qn, acc);
  nnclr_finalize_kernel<<<dim3(1), dim3(32), 0, stream>>>(acc, out);
}
